// GAT_21268678050526
// MI455X (gfx1250) — matrix operand order repaired and checked by reading, not compiled, not run on hardware
//
#include <hip/hip_runtime.h>
#include <hip/hip_bf16.h>

// ---------------------------------------------------------------------------
// Types for WMMA (gfx1250, wave32)
// ---------------------------------------------------------------------------
typedef float v8f __attribute__((ext_vector_type(8)));
typedef __bf16 bf16x16 __attribute__((ext_vector_type(16)));
typedef unsigned short u16x16 __attribute__((ext_vector_type(16)));

#define NEG_SLOPE 0.2f

// ---------------------------------------------------------------------------
// Helpers
// ---------------------------------------------------------------------------
__device__ __forceinline__ unsigned short f32_to_bf16_rne(float f) {
    unsigned u = __float_as_uint(f);
    u += 0x7FFFu + ((u >> 16) & 1u);
    return (unsigned short)(u >> 16);
}
__device__ __forceinline__ float bf16_bits_to_f32(unsigned short h) {
    return __uint_as_float(((unsigned)h) << 16);
}
// Monotone float->uint key so unsigned atomicMax == float max
__device__ __forceinline__ unsigned enc_key(float f) {
    unsigned u = __float_as_uint(f);
    return (u & 0x80000000u) ? ~u : (u | 0x80000000u);
}
__device__ __forceinline__ float dec_key(unsigned k) {
    return __uint_as_float((k & 0x80000000u) ? (k & 0x7FFFFFFFu) : ~k);
}
#define KEY_NEG_INF 0x007FFFFFu   // enc_key(-inf)

__device__ __forceinline__ float lrelu(float v) { return v > 0.f ? v : NEG_SLOPE * v; }

// ---------------------------------------------------------------------------
// Init: zero accumulators, set segment-max keys to -inf
// ---------------------------------------------------------------------------
__global__ void init_ws_kernel(float* __restrict__ act1, float* __restrict__ dout,
                               unsigned* __restrict__ k1, float* __restrict__ den1,
                               unsigned* __restrict__ k2, float* __restrict__ den2, int N) {
    int i = blockIdx.x * blockDim.x + threadIdx.x;
    if (i < N * 128) act1[i] = 0.f;
    if (i < N * 32)  dout[i] = 0.f;
    if (i < N * 4)   { k1[i] = KEY_NEG_INF; den1[i] = 0.f; }
    if (i < N)       { k2[i] = KEY_NEG_INF; den2[i] = 0.f; }
}

// ---------------------------------------------------------------------------
// Pack B[K,Nout] (fp32) into fragment-major bf16 hi/lo:
// record = (k0/32)*tilesN + tn ; per record 32 lanes x 16 bf16 contiguous.
// Lane layout (contiguous k order, same as the A fragment in the GEMM): elem e -> K = k0 + (lane>=16)*16 + e,
// col = tn*16 + lane%16 (ISA 7.12.2).
// ---------------------------------------------------------------------------
__global__ void pack_B_bf16(const float* __restrict__ B, unsigned short* __restrict__ hi,
                            unsigned short* __restrict__ lo, int K, int Nout) {
    int idx = blockIdx.x * blockDim.x + threadIdx.x;
    int tilesN = Nout >> 4;
    int total = (K >> 5) * tilesN * 32;
    if (idx >= total) return;
    int lane = idx & 31;
    int rec  = idx >> 5;                 // kblock * tilesN + tn
    int tn   = rec % tilesN;
    int k0   = (rec / tilesN) << 5;
    int half = lane >> 4, l15 = lane & 15;
    int col  = tn * 16 + l15;
    int kb   = k0 + half * 16;
    unsigned short* ph = hi + (size_t)idx * 16;
    unsigned short* pl = lo + (size_t)idx * 16;
#pragma unroll
    for (int e = 0; e < 16; ++e) {
        float f = B[(size_t)(kb + e) * Nout + col];
        unsigned short hs = f32_to_bf16_rne(f);
        ph[e] = hs;
        pl[e] = f32_to_bf16_rne(f - bf16_bits_to_f32(hs));
    }
}

// ---------------------------------------------------------------------------
// Blocked GEMM: C[M, TN*16] = A[M,K] @ B. One wave computes a full 16-row
// strip (16 x TN*16): A fragment converted once per k-step, reused across TN
// tiles; B fragments come prepacked (pure b128 loads). fp32 accuracy via
// bf16 hi/lo 3-product WMMA. K % 32 == 0.
// ---------------------------------------------------------------------------
template <int TN>
__global__ void gemm_wmma_bf16x3_blk(const float* __restrict__ A,
                                     const unsigned short* __restrict__ Bh,
                                     const unsigned short* __restrict__ Bl,
                                     float* __restrict__ C, int M, int K, int strips) {
    constexpr int NOUT = TN * 16;
    const int wid  = threadIdx.x >> 5;
    const int lane = threadIdx.x & 31;
    const int strip = blockIdx.x * (blockDim.x >> 5) + wid;
    if (strip >= strips) return;

    const int half = lane >> 4;        // 0: lanes 0-15, 1: lanes 16-31
    const int l15  = lane & 15;
    const bool full = (strip * 16 + 16) <= M;   // wave-uniform
    int row = strip * 16 + l15;
    const int rowc = full ? row : (row < M ? row : M - 1);

    v8f acc[TN];
#pragma unroll
    for (int t = 0; t < TN; ++t) acc[t] = (v8f){0.f, 0.f, 0.f, 0.f, 0.f, 0.f, 0.f, 0.f};

    for (int k0 = 0; k0 < K; k0 += 32) {
        // ---- A fragment: elems 0..7 -> K = k0 + half*16 + e ;
        //                  elems 8..15 -> K = k0 + half*16 + 8 + e
        const float* Ar = A + (size_t)rowc * K + k0;
        float af[16];
        const int b0 = half * 16, b1 = half * 16 + 8;
#pragma unroll
        for (int e = 0; e < 8; ++e) { af[e] = Ar[b0 + e]; af[8 + e] = Ar[b1 + e]; }
        u16x16 ah, al;
#pragma unroll
        for (int e = 0; e < 16; ++e) {
            unsigned short hs = f32_to_bf16_rne(af[e]);
            ah[e] = hs;
            al[e] = f32_to_bf16_rne(af[e] - bf16_bits_to_f32(hs));
        }
        bf16x16 vah = __builtin_bit_cast(bf16x16, ah);
        bf16x16 val = __builtin_bit_cast(bf16x16, al);

        const int recBase = (k0 >> 5) * TN;
#pragma unroll
        for (int tn = 0; tn < TN; ++tn) {
            size_t frag = ((size_t)(recBase + tn) * 32 + lane) * 16;
            u16x16 bhu = *(const u16x16*)(Bh + frag);
            u16x16 blu = *(const u16x16*)(Bl + frag);
            bf16x16 vbh = __builtin_bit_cast(bf16x16, bhu);
            bf16x16 vbl = __builtin_bit_cast(bf16x16, blu);
            acc[tn] = __builtin_amdgcn_wmma_f32_16x16x32_bf16(false, vah, false, vbh, (short)0, acc[tn], false, false);
            acc[tn] = __builtin_amdgcn_wmma_f32_16x16x32_bf16(false, vah, false, vbl, (short)0, acc[tn], false, false);
            acc[tn] = __builtin_amdgcn_wmma_f32_16x16x32_bf16(false, val, false, vbh, (short)0, acc[tn], false, false);
        }
    }

    // C/D layout: VGPR r, lane<16 -> M = r ; lane>=16 -> M = 8+r ; N = lane%16
    float* Cb = C + (size_t)(strip * 16 + half * 8) * NOUT + l15;
    if (full) {
#pragma unroll
        for (int r = 0; r < 8; ++r)
#pragma unroll
            for (int tn = 0; tn < TN; ++tn)
                Cb[(size_t)r * NOUT + tn * 16] = acc[tn][r];
    } else {
#pragma unroll
        for (int r = 0; r < 8; ++r) {
            if (strip * 16 + half * 8 + r < M)
#pragma unroll
                for (int tn = 0; tn < TN; ++tn)
                    Cb[(size_t)r * NOUT + tn * 16] = acc[tn][r];
        }
    }
}

// ---------------------------------------------------------------------------
// Attention logits: alpha_s[n,h] = sum_c h[n,h*32+c]*a_src[h,c] (layer 1, H=4)
// One node per 128-thread block, one head per wave.
// ---------------------------------------------------------------------------
__global__ void alpha_heads4(const float* __restrict__ h, const float* __restrict__ asrc,
                             const float* __restrict__ adst, float* __restrict__ out_s,
                             float* __restrict__ out_d, int N) {
    int n = blockIdx.x;
    if (n >= N) return;
    int t = threadIdx.x;  // 0..127 : head = t>>5, channel = t&31
    float v = h[(size_t)n * 128 + t];
    float s = v * asrc[t];
    float d = v * adst[t];
#pragma unroll
    for (int off = 16; off > 0; off >>= 1) {
        s += __shfl_xor(s, off, 32);
        d += __shfl_xor(d, off, 32);
    }
    if ((t & 31) == 0) {
        int hd = t >> 5;
        out_s[n * 4 + hd] = s;
        out_d[n * 4 + hd] = d;
    }
}

// Layer 2 (H=1, 32 channels): one node per wave
__global__ void alpha_single(const float* __restrict__ h2, const float* __restrict__ asrc,
                             const float* __restrict__ adst, float* __restrict__ out_s,
                             float* __restrict__ out_d, int N) {
    int wid = threadIdx.x >> 5, lane = threadIdx.x & 31;
    int n = blockIdx.x * (blockDim.x >> 5) + wid;
    if (n >= N) return;
    float v = h2[(size_t)n * 32 + lane];
    float s = v * asrc[lane];
    float d = v * adst[lane];
#pragma unroll
    for (int off = 16; off > 0; off >>= 1) {
        s += __shfl_xor(s, off, 32);
        d += __shfl_xor(d, off, 32);
    }
    if (lane == 0) { out_s[n] = s; out_d[n] = d; }
}

// ---------------------------------------------------------------------------
// Edge softmax passes.  Edges e<E come from edge_index; e>=E are self-loops.
// Node arrays are float4/uint4-aligned (256B carve) -> b128 loads.
// ---------------------------------------------------------------------------
__global__ void edge_max4(const int* __restrict__ ei, int E, int EA,
                          const float4* __restrict__ as4, const float4* __restrict__ ad4,
                          unsigned* __restrict__ keys) {
    int e = blockIdx.x * blockDim.x + threadIdx.x;
    if (e >= EA) return;
    int s = e < E ? ei[e] : (e - E);
    int d = e < E ? ei[E + e] : (e - E);
    float4 A = as4[s], D = ad4[d];
    float av[4] = {A.x, A.y, A.z, A.w};
    float dv[4] = {D.x, D.y, D.z, D.w};
#pragma unroll
    for (int h = 0; h < 4; ++h)
        atomicMax(&keys[d * 4 + h], enc_key(lrelu(av[h] + dv[h])));
}

__global__ void edge_sum4(const int* __restrict__ ei, int E, int EA,
                          const float4* __restrict__ as4, const float4* __restrict__ ad4,
                          const uint4* __restrict__ keys4, float* __restrict__ den) {
    int e = blockIdx.x * blockDim.x + threadIdx.x;
    if (e >= EA) return;
    int s = e < E ? ei[e] : (e - E);
    int d = e < E ? ei[E + e] : (e - E);
    float4 A = as4[s], D = ad4[d];
    uint4  Kk = keys4[d];
    float av[4] = {A.x, A.y, A.z, A.w};
    float dv[4] = {D.x, D.y, D.z, D.w};
    unsigned kv[4] = {Kk.x, Kk.y, Kk.z, Kk.w};
#pragma unroll
    for (int h = 0; h < 4; ++h) {
        float v = lrelu(av[h] + dv[h]);
        atomicAdd(&den[d * 4 + h], __expf(v - dec_key(kv[h])));
    }
}

// One wave per (edge, head); lane = channel. Coalesced gather + coalesced atomics.
__global__ void edge_scatter_l1(const int* __restrict__ ei, int E, int EA,
                                const float* __restrict__ as, const float* __restrict__ ad,
                                const unsigned* __restrict__ keys, const float* __restrict__ den,
                                const float* __restrict__ h1, float* __restrict__ out1) {
    int wid = threadIdx.x >> 5, lane = threadIdx.x & 31;
    long long gw = (long long)blockIdx.x * (blockDim.x >> 5) + wid;
    if (gw >= (long long)EA * 4) return;
    int e = (int)(gw >> 2);
    int h = (int)(gw & 3);
    int s = e < E ? ei[e] : (e - E);
    int d = e < E ? ei[E + e] : (e - E);
    float v = lrelu(as[s * 4 + h] + ad[d * 4 + h]);
    float w = __expf(v - dec_key(keys[d * 4 + h])) / den[d * 4 + h];
    float hv = h1[(size_t)s * 128 + h * 32 + lane];
    atomicAdd(&out1[(size_t)d * 128 + h * 32 + lane], w * hv);
}

__global__ void edge_max1(const int* __restrict__ ei, int E, int EA,
                          const float* __restrict__ as, const float* __restrict__ ad,
                          unsigned* __restrict__ keys) {
    int e = blockIdx.x * blockDim.x + threadIdx.x;
    if (e >= EA) return;
    int s = e < E ? ei[e] : (e - E);
    int d = e < E ? ei[E + e] : (e - E);
    atomicMax(&keys[d], enc_key(lrelu(as[s] + ad[d])));
}

__global__ void edge_sum1(const int* __restrict__ ei, int E, int EA,
                          const float* __restrict__ as, const float* __restrict__ ad,
                          const unsigned* __restrict__ keys, float* __restrict__ den) {
    int e = blockIdx.x * blockDim.x + threadIdx.x;
    if (e >= EA) return;
    int s = e < E ? ei[e] : (e - E);
    int d = e < E ? ei[E + e] : (e - E);
    float v = lrelu(as[s] + ad[d]);
    atomicAdd(&den[d], __expf(v - dec_key(keys[d])));
}

// One wave per edge; lane = channel (32 channels)
__global__ void edge_scatter_l2(const int* __restrict__ ei, int E, int EA,
                                const float* __restrict__ as, const float* __restrict__ ad,
                                const unsigned* __restrict__ keys, const float* __restrict__ den,
                                const float* __restrict__ h2, float* __restrict__ out) {
    int wid = threadIdx.x >> 5, lane = threadIdx.x & 31;
    int e = blockIdx.x * (blockDim.x >> 5) + wid;
    if (e >= EA) return;
    int s = e < E ? ei[e] : (e - E);
    int d = e < E ? ei[E + e] : (e - E);
    float v = lrelu(as[s] + ad[d]);
    float w = __expf(v - dec_key(keys[d])) / den[d];
    atomicAdd(&out[(size_t)d * 32 + lane], w * h2[(size_t)s * 32 + lane]);
}

// ---------------------------------------------------------------------------
// Epilogues
// ---------------------------------------------------------------------------
__global__ void bias_elu_kernel(float* __restrict__ a, const float* __restrict__ b, int total) {
    int i = blockIdx.x * blockDim.x + threadIdx.x;
    if (i >= total) return;
    float v = a[i] + b[i & 127];
    a[i] = v > 0.f ? v : (__expf(v) - 1.f);
}

__global__ void bias_out_kernel(float* __restrict__ o, const float* __restrict__ b, int total) {
    int i = blockIdx.x * blockDim.x + threadIdx.x;
    if (i >= total) return;
    o[i] += b[i & 31];
}

// ---------------------------------------------------------------------------
// Launch
// ---------------------------------------------------------------------------
extern "C" void kernel_launch(void* const* d_in, const int* in_sizes, int n_in,
                              void* d_out, int out_size, void* d_ws, size_t ws_size,
                              hipStream_t stream) {
    const float* x   = (const float*)d_in[0];
    const int*   ei  = (const int*)d_in[1];
    const float* W1  = (const float*)d_in[2];
    const float* as1 = (const float*)d_in[3];
    const float* ad1 = (const float*)d_in[4];
    const float* b1  = (const float*)d_in[5];
    const float* W2  = (const float*)d_in[6];
    const float* as2 = (const float*)d_in[7];
    const float* ad2 = (const float*)d_in[8];
    const float* b2  = (const float*)d_in[9];
    float* out = (float*)d_out;

    const int N  = in_sizes[0] / 256;   // IN_CH = 256
    const int E  = in_sizes[1] / 2;
    const int EA = E + N;               // + self loops

    // Workspace carving (256B aligned)
    size_t off = 0;
    auto carve = [&](size_t bytes) -> void* {
        void* p = (char*)d_ws + off;
        off += (bytes + 255) & ~(size_t)255;
        return p;
    };
    float*    h1    = (float*)carve((size_t)N * 128 * 4);
    float*    act1  = (float*)carve((size_t)N * 128 * 4);   // layer-1 scatter accum -> ELU
    float*    as1v  = (float*)carve((size_t)N * 4 * 4);
    float*    ad1v  = (float*)carve((size_t)N * 4 * 4);
    unsigned* k1    = (unsigned*)carve((size_t)N * 4 * 4);
    float*    den1  = (float*)carve((size_t)N * 4 * 4);
    float*    h2    = (float*)carve((size_t)N * 32 * 4);
    float*    as2v  = (float*)carve((size_t)N * 4);
    float*    ad2v  = (float*)carve((size_t)N * 4);
    unsigned* k2    = (unsigned*)carve((size_t)N * 4);
    float*    den2  = (float*)carve((size_t)N * 4);
    // packed weights (fragment-major bf16 hi/lo)
    const int recs1 = (256 / 32) * (128 / 16);   // 64 records
    const int recs2 = (128 / 32) * (32 / 16);    // 8 records
    unsigned short* B1h = (unsigned short*)carve((size_t)recs1 * 32 * 16 * 2);
    unsigned short* B1l = (unsigned short*)carve((size_t)recs1 * 32 * 16 * 2);
    unsigned short* B2h = (unsigned short*)carve((size_t)recs2 * 32 * 16 * 2);
    unsigned short* B2l = (unsigned short*)carve((size_t)recs2 * 32 * 16 * 2);
    (void)ws_size; (void)n_in; (void)out_size;

    const int TPB = 256;

    // 0. zero accumulators / init segment-max keys; pack weights
    init_ws_kernel<<<(N * 128 + TPB - 1) / TPB, TPB, 0, stream>>>(act1, out, k1, den1, k2, den2, N);
    pack_B_bf16<<<(recs1 * 32 + TPB - 1) / TPB, TPB, 0, stream>>>(W1, B1h, B1l, 256, 128);
    pack_B_bf16<<<(recs2 * 32 + TPB - 1) / TPB, TPB, 0, stream>>>(W2, B2h, B2l, 128, 32);

    // 1. Layer-1 GEMM: h1 = x @ W1   (M=N, K=256, Nout=128, TN=8)
    {
        int strips = (N + 15) / 16;
        gemm_wmma_bf16x3_blk<8><<<(strips + 7) / 8, TPB, 0, stream>>>(x, B1h, B1l, h1, N, 256, strips);
    }

    // 2. attention logits per node/head
    alpha_heads4<<<N, 128, 0, stream>>>(h1, as1, ad1, as1v, ad1v, N);

    // 3. per-dst softmax over incoming edges (3 passes) + weighted scatter
    edge_max4<<<(EA + TPB - 1) / TPB, TPB, 0, stream>>>(ei, E, EA, (const float4*)as1v,
                                                        (const float4*)ad1v, k1);
    edge_sum4<<<(EA + TPB - 1) / TPB, TPB, 0, stream>>>(ei, E, EA, (const float4*)as1v,
                                                        (const float4*)ad1v, (const uint4*)k1, den1);
    {
        long long waves = (long long)EA * 4;
        int blocks = (int)((waves + 7) / 8);
        edge_scatter_l1<<<blocks, TPB, 0, stream>>>(ei, E, EA, as1v, ad1v, k1, den1, h1, act1);
    }

    // 4. + b1, ELU
    bias_elu_kernel<<<(N * 128 + TPB - 1) / TPB, TPB, 0, stream>>>(act1, b1, N * 128);

    // 5. Layer-2 GEMM: h2 = act1 @ W2  (M=N, K=128, Nout=32, TN=2)
    {
        int strips = (N + 15) / 16;
        gemm_wmma_bf16x3_blk<2><<<(strips + 7) / 8, TPB, 0, stream>>>(act1, B2h, B2l, h2, N, 128, strips);
    }

    // 6. layer-2 attention + softmax + scatter into d_out
    alpha_single<<<(N + 7) / 8, TPB, 0, stream>>>(h2, as2, ad2, as2v, ad2v, N);
    edge_max1<<<(EA + TPB - 1) / TPB, TPB, 0, stream>>>(ei, E, EA, as2v, ad2v, k2);
    edge_sum1<<<(EA + TPB - 1) / TPB, TPB, 0, stream>>>(ei, E, EA, as2v, ad2v, k2, den2);
    edge_scatter_l2<<<(EA + 7) / 8, TPB, 0, stream>>>(ei, E, EA, as2v, ad2v, k2, den2, h2, out);

    // 7. + b2
    bias_out_kernel<<<(N * 32 + TPB - 1) / TPB, TPB, 0, stream>>>(out, b2, N * 32);
}
